// MessagePassingLayer_62732292325695
// MI455X (gfx1250) — matrix operand order repaired and checked by reading, not compiled, not run on hardware
//
#include <hip/hip_runtime.h>
#include <hip/hip_bf16.h>

// ---------------------------------------------------------------- types
typedef __attribute__((ext_vector_type(8)))  __bf16 v8bf;
typedef __attribute__((ext_vector_type(16))) __bf16 v16bf;
typedef __attribute__((ext_vector_type(8)))  float  v8f;
typedef __attribute__((ext_vector_type(4)))  int    v4i;

#define BN   2048    // B*N rows
#define NN   128     // N
#define DD   128     // D
#define EE   128     // E
#define HH   256     // H

#if __has_builtin(__builtin_amdgcn_global_load_async_to_lds_b128) && \
    __has_builtin(__builtin_amdgcn_s_wait_asynccnt)
#define USE_ASYNC_LDS 1
typedef __attribute__((address_space(1))) v4i* gv4i_ptr;   // global src
typedef __attribute__((address_space(3))) v4i* lv4i_ptr;   // LDS dst
#else
#define USE_ASYNC_LDS 0
#endif

__device__ __forceinline__ v16bf cat8(v8bf lo, v8bf hi) {
  return __builtin_shufflevector(lo, hi, 0,1,2,3,4,5,6,7,8,9,10,11,12,13,14,15);
}

// A-fragment (16x32 bf16, MxK) per ISA 7.12.2:
// lanes 0-15: row M=lane, K = k0+[0..7] and k0+[16..23]
// lanes 16-31: row M=lane-16, K = k0+[8..15] and k0+[24..31]
__device__ __forceinline__ v16bf load_a_bf16(const __bf16* base, int m0, int k0,
                                             int ld, int lane) {
  int m  = m0 + (lane & 15);
  int kh = k0 + ((lane >> 4) << 3);
  const v8bf* p = (const v8bf*)(base + (size_t)m * ld + kh);
  return cat8(p[0], p[2]);     // kh..kh+7  and  kh+16..kh+23
}

// Same A-fragment from an fp32 source, converting to bf16 in-register.
__device__ __forceinline__ v16bf load_a_f32(const float* base, int m0, int k0,
                                            int ld, int lane) {
  int m  = m0 + (lane & 15);
  int kh = k0 + ((lane >> 4) << 3);
  const float* r = base + (size_t)m * ld + kh;
  v16bf a;
#pragma unroll
  for (int t = 0; t < 8; ++t) { a[t] = (__bf16)r[t]; a[t + 8] = (__bf16)r[t + 16]; }
  return a;
}

// B-fragment (32x16 bf16, KxN), source stored column-major BT[n][k] (ld = K stride):
// lanes 0-15: col N=lane, K = k0+[0..7] and k0+[16..23]; lanes 16-31: col N=lane-16, K = k0+[8..15] and k0+[24..31]
__device__ __forceinline__ v16bf load_b_bf16(const __bf16* baseT, int n0, int k0,
                                             int ld, int lane) {
  int n  = n0 + (lane & 15);
  int kk = k0 + ((lane >> 4) << 3);
  const v8bf* p = (const v8bf*)(baseT + (size_t)n * ld + kk);
  return cat8(p[0], p[2]);
}

__device__ __forceinline__ v8f wmma_bf16(v16bf a, v16bf b, v8f c) {
  return __builtin_amdgcn_wmma_f32_16x16x32_bf16(false, a, false, b,
                                                 (short)0, c, false, false);
}

// ---------------------------------------------------------------- prep kernels
// dst[n*K + k] = bf16(src[(k+rowOff)*ld + n])   (transpose + convert)
__global__ void k_transpose_bf16(const float* __restrict__ src, __bf16* __restrict__ dst,
                                 int K, int N, int rowOff, int ld) {
  int idx = blockIdx.x * blockDim.x + threadIdx.x;
  if (idx >= K * N) return;
  int k = idx % K, n = idx / K;
  dst[(size_t)n * K + k] = (__bf16)src[(size_t)(k + rowOff) * ld + n];
}

__global__ void k_cvt_bf16(const float* __restrict__ src, __bf16* __restrict__ dst, int n) {
  int idx = blockIdx.x * blockDim.x + threadIdx.x;
  if (idx < n) dst[idx] = (__bf16)src[idx];
}

__global__ void k_cnt(const unsigned char* __restrict__ mask, float* __restrict__ cnt) {
  int b = threadIdx.x;
  if (b < 16) {
    float s = 0.f;
    for (int j = 0; j < NN; ++j) s += mask[b * NN + j] ? 1.f : 0.f;
    cnt[b] = s;
  }
}

// ---------------------------------------------------------------- p = nodes @ W1[:D] + b1
__global__ __launch_bounds__(256) void k_p(const __bf16* __restrict__ nodesbf,
                                           const __bf16* __restrict__ W1nT,
                                           const float* __restrict__ b1,
                                           float* __restrict__ p) {
  int blk  = blockIdx.x;             // 16-row tile of BN
  int wave = threadIdx.x >> 5, lane = threadIdx.x & 31;
  int m0 = blk << 4;
  v16bf a[4];
#pragma unroll
  for (int kk = 0; kk < 4; ++kk) a[kk] = load_a_bf16(nodesbf, m0, kk * 32, DD, lane);
#pragma unroll
  for (int t = 0; t < 2; ++t) {
    int n0 = (wave * 2 + t) << 4;
    v8f c = {};
#pragma unroll
    for (int kk = 0; kk < 4; ++kk)
      c = wmma_bf16(a[kk], load_b_bf16(W1nT, n0, kk * 32, DD, lane), c);
    int col = n0 + (lane & 15);
    int rhi = (lane >> 4) << 3;
    float bb = b1[col];
#pragma unroll
    for (int r = 0; r < 8; ++r) {
      int row = m0 + r + rhi;
      p[(size_t)row * HH + col] = c[r] + bb;
    }
  }
}

// ---------------------------------------------------------------- main edge kernel
// one block per (b,i): s[h] = sum_j m_j * relu(p_i[h] + (E_{b,i} @ W1e)[j,h])
__global__ __launch_bounds__(256) void k_edge_msg(const float* __restrict__ edges,
                                                  const float* __restrict__ p,
                                                  const unsigned char* __restrict__ mask,
                                                  const __bf16* __restrict__ W1eT,
                                                  __bf16* __restrict__ S) {
  __shared__ __bf16 ldsW[HH * 136];   // W1e^T, padded stride: conflict-free ds_load_b128
  __shared__ float  ldsP[HH];
  __shared__ float  ldsM[NN];
  __shared__ float  ldsS[HH];

  int blk = blockIdx.x;               // b*N + i
  int b   = blk >> 7;
  int tid = threadIdx.x;

#if USE_ASYNC_LDS
  { // stage W1e^T [256 h][128 k] -> LDS via async DMA (ASYNCcnt path), one row/thread
    char* gsrc = (char*)(void*)(W1eT + (size_t)tid * EE);
    char* ldst = (char*)(ldsW + tid * 136);
#pragma unroll
    for (int t = 0; t < 16; ++t) {
      __builtin_amdgcn_global_load_async_to_lds_b128(
          (gv4i_ptr)(void*)(gsrc + t * 16), (lv4i_ptr)(void*)(ldst + t * 16), 0, 0);
    }
  }
#else
  { // fallback: stage through VGPRs
    const v8bf* src = (const v8bf*)(W1eT + (size_t)tid * EE);
    v8bf* dst = (v8bf*)(ldsW + tid * 136);
#pragma unroll
    for (int t = 0; t < 16; ++t) dst[t] = src[t];
  }
#endif
  ldsP[tid] = p[(size_t)blk * HH + tid];
  if (tid < NN) ldsM[tid] = mask[b * NN + tid] ? 1.f : 0.f;
  ldsS[tid] = 0.f;

  // overlap: pull this wave's edge A-fragments (fp32 -> bf16) while DMA runs
  int wave = tid >> 5, lane = tid & 31;
  int m0 = wave << 4;                  // this wave's 16-row j strip
  const float* Eb = edges + (size_t)blk * (NN * EE);

  v16bf a[4];
#pragma unroll
  for (int kk = 0; kk < 4; ++kk) a[kk] = load_a_f32(Eb, m0, kk * 32, EE, lane);

#if USE_ASYNC_LDS
  __builtin_amdgcn_s_wait_asynccnt(0);
#endif
  __syncthreads();

  int rhi = (lane >> 4) << 3;
#pragma unroll 4
  for (int nt = 0; nt < 16; ++nt) {
    int n0 = nt << 4;
    v8f c = {};
#pragma unroll
    for (int kk = 0; kk < 4; ++kk)
      c = wmma_bf16(a[kk], load_b_bf16(ldsW, n0, kk * 32, 136, lane), c);
    int col = n0 + (lane & 15);
    float pc = ldsP[col];
    float partial = 0.f;
#pragma unroll
    for (int r = 0; r < 8; ++r) {
      int row = m0 + r + rhi;
      float v = c[r] + pc;
      v = v > 0.f ? v : 0.f;           // relu
      partial += v * ldsM[row];        // sender mask m_j
    }
    atomicAdd(&ldsS[col], partial);    // ds_add_f32 column reduction
  }
  __syncthreads();
  S[(size_t)blk * HH + tid] = (__bf16)ldsS[tid];
}

// ---------------------------------------------------------------- agg = S @ W2 + cnt*b2
__global__ __launch_bounds__(256) void k_agg(const __bf16* __restrict__ S,
                                             const __bf16* __restrict__ W2T,
                                             const float* __restrict__ b2,
                                             const float* __restrict__ cnt,
                                             __bf16* __restrict__ agg) {
  int blk  = blockIdx.x;
  int wave = threadIdx.x >> 5, lane = threadIdx.x & 31;
  int m0 = blk << 4;
  v16bf a[8];
#pragma unroll
  for (int kk = 0; kk < 8; ++kk) a[kk] = load_a_bf16(S, m0, kk * 32, HH, lane);
#pragma unroll
  for (int t = 0; t < 2; ++t) {
    int n0 = (wave * 2 + t) << 4;
    v8f c = {};
#pragma unroll
    for (int kk = 0; kk < 8; ++kk)
      c = wmma_bf16(a[kk], load_b_bf16(W2T, n0, kk * 32, HH, lane), c);
    int col = n0 + (lane & 15);
    int rhi = (lane >> 4) << 3;
    float bb = b2[col];
#pragma unroll
    for (int r = 0; r < 8; ++r) {
      int row = m0 + r + rhi;
      agg[(size_t)row * HH + col] = (__bf16)(c[r] + cnt[row >> 7] * bb);
    }
  }
}

// ---------------------------------------------------------------- u = relu([nodes,agg]@U1 + c1)
__global__ __launch_bounds__(256) void k_u(const __bf16* __restrict__ nodesbf,
                                           const __bf16* __restrict__ agg,
                                           const __bf16* __restrict__ U1T,
                                           const float* __restrict__ c1,
                                           __bf16* __restrict__ u) {
  int blk  = blockIdx.x;
  int wave = threadIdx.x >> 5, lane = threadIdx.x & 31;
  int m0 = blk << 4;
  v16bf a[12];                          // K = D + H = 384
#pragma unroll
  for (int kk = 0; kk < 4; ++kk)  a[kk] = load_a_bf16(nodesbf, m0, kk * 32, DD, lane);
#pragma unroll
  for (int kk = 4; kk < 12; ++kk) a[kk] = load_a_bf16(agg, m0, (kk - 4) * 32, HH, lane);
#pragma unroll
  for (int t = 0; t < 2; ++t) {
    int n0 = (wave * 2 + t) << 4;
    v8f c = {};
#pragma unroll
    for (int kk = 0; kk < 12; ++kk)
      c = wmma_bf16(a[kk], load_b_bf16(U1T, n0, kk * 32, 384, lane), c);
    int col = n0 + (lane & 15);
    int rhi = (lane >> 4) << 3;
    float bb = c1[col];
#pragma unroll
    for (int r = 0; r < 8; ++r) {
      int row = m0 + r + rhi;
      float v = c[r] + bb;
      u[(size_t)row * HH + col] = (__bf16)(v > 0.f ? v : 0.f);
    }
  }
}

// -------------------------------------------- upd = u@U2+c2 ; residual + LayerNorm + mask
__global__ __launch_bounds__(256) void k_final(const __bf16* __restrict__ u,
                                               const __bf16* __restrict__ U2T,
                                               const float* __restrict__ c2,
                                               const float* __restrict__ nodes,
                                               const unsigned char* __restrict__ mask,
                                               const float* __restrict__ gamma,
                                               const float* __restrict__ beta,
                                               float* __restrict__ out) {
  __shared__ float xb[16][132];         // padded
  __shared__ float mu_s[16], rs_s[16];
  int blk  = blockIdx.x;                // 16 rows
  int tid  = threadIdx.x;
  int wave = tid >> 5, lane = tid & 31;
  int m0 = blk << 4;

  v16bf a[8];
#pragma unroll
  for (int kk = 0; kk < 8; ++kk) a[kk] = load_a_bf16(u, m0, kk * 32, HH, lane);
  int n0 = wave << 4;                   // 8 waves cover N = D = 128
  v8f c = {};
#pragma unroll
  for (int kk = 0; kk < 8; ++kk)
    c = wmma_bf16(a[kk], load_b_bf16(U2T, n0, kk * 32, HH, lane), c);

  int col = n0 + (lane & 15);
  int rhi = (lane >> 4) << 3;
  float cc = c2[col];
#pragma unroll
  for (int r = 0; r < 8; ++r) {
    int lr  = r + rhi;
    int row = m0 + lr;
    float mf = mask[row] ? 1.f : 0.f;
    float upd = (c[r] + cc) * mf;       // updates * mask_i
    xb[lr][col] = nodes[(size_t)row * DD + col] + upd;
  }
  __syncthreads();

  if (tid < 16) {                       // LayerNorm stats per row
    float mu = 0.f;
    for (int k = 0; k < DD; ++k) mu += xb[tid][k];
    mu *= (1.f / DD);
    float var = 0.f;
    for (int k = 0; k < DD; ++k) { float d = xb[tid][k] - mu; var += d * d; }
    var *= (1.f / DD);
    mu_s[tid] = mu;
    rs_s[tid] = rsqrtf(var + 1e-5f);
  }
  __syncthreads();

  for (int e = tid; e < 16 * DD; e += 256) {
    int lr = e >> 7, k = e & 127;
    int row = m0 + lr;
    float mf = mask[row] ? 1.f : 0.f;
    float xn = (xb[lr][k] - mu_s[lr]) * rs_s[lr];
    out[(size_t)row * DD + k] = (xn * gamma[k] + beta[k]) * mf;
  }
}

// ---------------------------------------------------------------- launcher
extern "C" void kernel_launch(void* const* d_in, const int* in_sizes, int n_in,
                              void* d_out, int out_size, void* d_ws, size_t ws_size,
                              hipStream_t stream) {
  const float* nodes = (const float*)d_in[0];
  const float* edges = (const float*)d_in[1];
  const unsigned char* mask = (const unsigned char*)d_in[2];
  const float* W1 = (const float*)d_in[3];
  const float* b1 = (const float*)d_in[4];
  const float* W2 = (const float*)d_in[5];
  const float* b2 = (const float*)d_in[6];
  const float* U1 = (const float*)d_in[7];
  const float* c1 = (const float*)d_in[8];
  const float* U2 = (const float*)d_in[9];
  const float* c2 = (const float*)d_in[10];
  const float* gamma = (const float*)d_in[11];
  const float* beta = (const float*)d_in[12];
  float* out = (float*)d_out;

  char* ws = (char*)d_ws;
  size_t off = 0;
  auto alloc = [&](size_t bytes) -> char* {
    char* q = ws + off;
    off = (off + bytes + 255) & ~(size_t)255;
    return q;
  };
  __bf16* W1nT    = (__bf16*)alloc((size_t)HH * DD * 2);   // [h][k<128]
  __bf16* W1eT    = (__bf16*)alloc((size_t)HH * EE * 2);   // [h][k] of W1[128:,:]
  __bf16* W2T     = (__bf16*)alloc((size_t)HH * HH * 2);
  __bf16* U1T     = (__bf16*)alloc((size_t)HH * 384 * 2);
  __bf16* U2T     = (__bf16*)alloc((size_t)DD * HH * 2);
  __bf16* nodesbf = (__bf16*)alloc((size_t)BN * DD * 2);
  float*  p       = (float*) alloc((size_t)BN * HH * 4);
  __bf16* S       = (__bf16*)alloc((size_t)BN * HH * 2);
  __bf16* agg     = (__bf16*)alloc((size_t)BN * HH * 2);
  __bf16* uu      = (__bf16*)alloc((size_t)BN * HH * 2);
  float*  cnt     = (float*) alloc(16 * 4);
  if (off > ws_size) return;   // workspace too small: refuse to launch

  // prep: weight transposes + conversions
  k_transpose_bf16<<<(DD * HH + 255) / 256, 256, 0, stream>>>(W1, W1nT, DD, HH, 0,   HH);
  k_transpose_bf16<<<(EE * HH + 255) / 256, 256, 0, stream>>>(W1, W1eT, EE, HH, DD,  HH);
  k_transpose_bf16<<<(HH * HH + 255) / 256, 256, 0, stream>>>(W2, W2T,  HH, HH, 0,   HH);
  k_transpose_bf16<<<(384 * HH + 255) / 256, 256, 0, stream>>>(U1, U1T, 384, HH, 0,  HH);
  k_transpose_bf16<<<(HH * DD + 255) / 256, 256, 0, stream>>>(U2, U2T,  HH, DD, 0,   DD);
  k_cvt_bf16<<<(BN * DD + 255) / 256, 256, 0, stream>>>(nodes, nodesbf, BN * DD);
  k_cnt<<<1, 16, 0, stream>>>(mask, cnt);

  // p_i = nodes @ W1[:D] + b1  (hoisted receiver contribution)
  k_p<<<BN / 16, 256, 0, stream>>>(nodesbf, W1nT, b1, p);

  // main: per-(b,i) edge GEMM with fused relu/mask/column-sum
  k_edge_msg<<<BN, 256, 0, stream>>>(edges, p, mask, W1eT, S);

  // agg = S @ W2 + cnt*b2   (second linear pushed through the j-sum)
  k_agg<<<BN / 16, 256, 0, stream>>>(S, W2T, b2, cnt, agg);

  // update MLP
  k_u<<<BN / 16, 256, 0, stream>>>(nodesbf, agg, U1T, c1, uu);

  // final projection + residual + LayerNorm + mask
  k_final<<<BN / 16, 256, 0, stream>>>(uu, U2T, c2, nodes, mask, gamma, beta, out);
}
